// GRU_53850299957520
// MI455X (gfx1250) — matrix operand order repaired and checked by reading, not compiled, not run on hardware
//
#include <hip/hip_runtime.h>
#include <math.h>

typedef __attribute__((ext_vector_type(16))) __bf16 v16bf;
typedef __attribute__((ext_vector_type(8)))  float  v8f;

#define NCOLS 1536   // 3*H
#define NT    96     // NCOLS / 16

__device__ __forceinline__ unsigned short f2bf(float f) {
  unsigned int u = __float_as_uint(f);
  unsigned int r = 0x7FFFu + ((u >> 16) & 1u);   // round-to-nearest-even
  return (unsigned short)((u + r) >> 16);
}
__device__ __forceinline__ float bf2f(unsigned short h) {
  return __uint_as_float(((unsigned int)h) << 16);
}

union FragBF16 {
  uint4 q[2];
  v16bf v;
};

// ---------------- f32 -> bf16 bulk convert ----------------
__global__ void cvt_f32_to_bf16(const float* __restrict__ src,
                                unsigned short* __restrict__ dst, int n) {
  int i = blockIdx.x * blockDim.x + threadIdx.x;
  if (i < n) dst[i] = f2bf(src[i]);
}

// ---------------- pack W (K x 1536, row-major f32) into WMMA-B fragment order ----------------
// Packed index: ((kt*NT + nt)*32 + lane)*16 + e
// B layout (v_wmma_*_16x16x32): lane holds column n = nt*16 + (lane&15),
// K = kt*32 + (lane>>4)*8 + (e&7) + (e>>3)*16   (two 8-K chunks per lane)
__global__ void pack_b(const float* __restrict__ W,
                       unsigned short* __restrict__ P, int Ktot) {
  int idx = blockIdx.x * blockDim.x + threadIdx.x;      // < Ktot*1536, exact grid
  int e    = idx & 15;
  int lane = (idx >> 4) & 31;
  int tile = idx >> 9;
  int nt = tile % NT;
  int kt = tile / NT;
  int n = nt * 16 + (lane & 15);
  int k = kt * 32 + ((lane >> 4) << 3) + (e & 7) + ((e >> 3) << 4);
  P[idx] = f2bf(W[(size_t)k * NCOLS + n]);
}

// ---------------- zero hidden state (f32 + bf16 copies) ----------------
__global__ void zero_h(float* __restrict__ hf, unsigned short* __restrict__ hb, int n) {
  int i = blockIdx.x * blockDim.x + threadIdx.x;
  if (i < n) { hf[i] = 0.0f; hb[i] = 0; }
}

// ---------------- WMMA GEMM: C(M x 1536) = A(M x K, bf16 row-major) * Bpacked ----------------
// Register-blocked: each wave computes an (MREP*16) x (NREP*16) tile, reusing A-fragments
// across NREP WMMAs and B-fragments across MREP WMMAs. 4 waves / 128-thread block,
// grids divide exactly -> EXEC all ones around every WMMA (ISA requirement).
template<int KTILES, int MREP, int NREP, bool STORE_BF16>
__global__ void wmma_gemm_n1536(const unsigned short* __restrict__ A,
                                const unsigned short* __restrict__ Bp,
                                void* __restrict__ Cout) {
  const int lane = threadIdx.x & 31;
  const int wave = threadIdx.x >> 5;
  const int group = blockIdx.x * 4 + wave;
  const int NGRP = NT / NREP;
  const int mg = group / NGRP;
  const int ng = group - mg * NGRP;
  const int mt0 = mg * MREP;
  const int nt0 = ng * NREP;
  const int lda = KTILES * 32;

  // A fragment addressing (16-bit A 16x32 layout): two contiguous 16B chunks per lane
  const int mrowLane = lane & 15;
  const int koff = (lane >> 4) << 3;

  const unsigned short* aBase[MREP];
#pragma unroll
  for (int mi = 0; mi < MREP; ++mi)
    aBase[mi] = A + (size_t)((mt0 + mi) * 16 + mrowLane) * lda + koff;

  // B fragment addressing (pre-packed, 32B contiguous per lane per k-tile)
  const unsigned short* bBase[NREP];
#pragma unroll
  for (int ni = 0; ni < NREP; ++ni)
    bBase[ni] = Bp + ((size_t)(nt0 + ni) * 32 + lane) * 16;

  v8f c[MREP][NREP] = {};

#pragma unroll
  for (int kt = 0; kt < KTILES; ++kt) {
    FragBF16 a[MREP], b[NREP];
#pragma unroll
    for (int mi = 0; mi < MREP; ++mi) {
      const unsigned short* ap = aBase[mi] + kt * 32;
      a[mi].q[0] = *(const uint4*)(ap);       // K = koff .. koff+7
      a[mi].q[1] = *(const uint4*)(ap + 16);  // K = koff+16 .. koff+23
    }
#pragma unroll
    for (int ni = 0; ni < NREP; ++ni) {
      const unsigned short* bp = bBase[ni] + (size_t)kt * (NT * 32 * 16);
      b[ni].q[0] = *(const uint4*)(bp);
      b[ni].q[1] = *(const uint4*)(bp + 8);
    }
#pragma unroll
    for (int mi = 0; mi < MREP; ++mi)
#pragma unroll
      for (int ni = 0; ni < NREP; ++ni)
        c[mi][ni] = __builtin_amdgcn_wmma_f32_16x16x32_bf16(
                        /*neg_a=*/false, a[mi].v, /*neg_b=*/false, b[ni].v,
                        /*c_mod=*/(short)0, c[mi][ni],
                        /*reuse_a=*/false, /*reuse_b=*/false);
  }

  // C/D layout: VGPR r -> M = (lane>=16 ? 8 : 0)+r, N = lane&15
  const int nlane = lane & 15;
  const int mhalf = (lane >> 4) << 3;
#pragma unroll
  for (int mi = 0; mi < MREP; ++mi) {
#pragma unroll
    for (int ni = 0; ni < NREP; ++ni) {
      const int ncol  = (nt0 + ni) * 16 + nlane;
      const int mrow0 = (mt0 + mi) * 16 + mhalf;
      if (STORE_BF16) {
        unsigned short* C = (unsigned short*)Cout;
#pragma unroll
        for (int r = 0; r < 8; ++r)
          C[(size_t)(mrow0 + r) * NCOLS + ncol] = f2bf(c[mi][ni][r]);
      } else {
        float* C = (float*)Cout;
#pragma unroll
        for (int r = 0; r < 8; ++r)
          C[(size_t)(mrow0 + r) * NCOLS + ncol] = c[mi][ni][r];
      }
    }
  }
}

// ---------------- fused GRU gate for one timestep ----------------
__global__ void gru_gate(const float* __restrict__ hg,          // B x 1536 (this step)
                         const unsigned short* __restrict__ xg, // (B*T) x 1536 bf16
                         const float* __restrict__ b_i,
                         const float* __restrict__ b_h,
                         float* __restrict__ hf,                // B x H f32 state
                         unsigned short* __restrict__ hb,       // B x H bf16 state
                         float* __restrict__ out,               // B x T x H
                         int t) {
  const int H = 512, T = 512;
  int idx = blockIdx.x * blockDim.x + threadIdx.x;  // < B*H, exact grid
  int b = idx >> 9;
  int j = idx & 511;
  size_t xrow = ((size_t)b * T + t) * NCOLS;
  size_t grow = (size_t)b * NCOLS;

  float xz = bf2f(xg[xrow + j])         + b_i[j];
  float xr = bf2f(xg[xrow + H + j])     + b_i[H + j];
  float xn = bf2f(xg[xrow + 2*H + j])   + b_i[2*H + j];
  float hz = hg[grow + j]        + b_h[j];
  float hr = hg[grow + H + j]    + b_h[H + j];
  float hnn = hg[grow + 2*H + j] + b_h[2*H + j];
  float hp = hf[idx];

  float z = 1.0f / (1.0f + __expf(-(xz + hz)));
  float r = 1.0f / (1.0f + __expf(-(xr + hr)));
  float n = tanhf(xn + r * hnn);
  float hnew = (1.0f - z) * n + z * hp;

  hf[idx] = hnew;
  hb[idx] = f2bf(hnew);
  out[((size_t)b * T + t) * H + j] = hnew;
}

extern "C" void kernel_launch(void* const* d_in, const int* in_sizes, int n_in,
                              void* d_out, int out_size, void* d_ws, size_t ws_size,
                              hipStream_t stream) {
  const float* x   = (const float*)d_in[0];
  const float* w_i = (const float*)d_in[1];
  const float* b_i = (const float*)d_in[2];
  const float* w_h = (const float*)d_in[3];
  const float* b_h = (const float*)d_in[4];
  float* out = (float*)d_out;

  const int B = 256, T = 512, D = 256, H = 512;

  // ---- workspace carving (all 256B aligned) ----
  char* ws = (char*)d_ws;
  size_t off = 0;
  auto carve = [&](size_t bytes) -> void* {
    void* p = ws + off;
    off = (off + bytes + 255) & ~(size_t)255;
    return p;
  };
  unsigned short* xg    = (unsigned short*)carve((size_t)B * T * NCOLS * 2); // 402 MB
  unsigned short* xb    = (unsigned short*)carve((size_t)B * T * D * 2);     // 67 MB
  unsigned short* wip   = (unsigned short*)carve((size_t)D * NCOLS * 2);
  unsigned short* whp   = (unsigned short*)carve((size_t)H * NCOLS * 2);
  float*          hgbuf = (float*)carve((size_t)B * NCOLS * 4);
  float*          hf    = (float*)carve((size_t)B * H * 4);
  unsigned short* hb    = (unsigned short*)carve((size_t)B * H * 2);
  (void)ws_size; (void)out_size; (void)in_sizes; (void)n_in;

  // ---- phase 0: precision demotion + weight packing + state init ----
  cvt_f32_to_bf16<<<(B * T * D) / 256, 256, 0, stream>>>(x, xb, B * T * D);
  pack_b<<<(D * NCOLS) / 256, 256, 0, stream>>>(w_i, wip, D);
  pack_b<<<(H * NCOLS) / 256, 256, 0, stream>>>(w_h, whp, H);
  zero_h<<<(B * H) / 256, 256, 0, stream>>>(hf, hb, B * H);

  // ---- phase 1: xg = x @ w_i  (M=131072, K=256), 2x4 register tile ----
  {
    const int groups = (B * T / 16 / 2) * (NT / 4);   // 4096 * 24 = 98304 waves
    wmma_gemm_n1536<8, 2, 4, true><<<groups / 4, 128, 0, stream>>>(xb, wip, xg);
  }

  // ---- phase 2: sequential recurrence (1x2 tile keeps 768 waves/step alive) ----
  for (int t = 0; t < T; ++t) {
    const int groups = (B / 16) * (NT / 2);           // 16 * 48 = 768 waves
    wmma_gemm_n1536<16, 1, 2, false><<<groups / 4, 128, 0, stream>>>(hb, whp, hgbuf);
    gru_gate<<<(B * H) / 256, 256, 0, stream>>>(hgbuf, xg, b_i, b_h, hf, hb, out, t);
  }
}
